// PaiNN_18923625906598
// MI455X (gfx1250) — matrix operand order repaired and checked by reading, not compiled, not run on hardware
//
#include <hip/hip_runtime.h>
#include <math.h>

// ---------------------------------------------------------------------------
// PaiNN for MI455X (gfx1250, wave32). GEMMs: v_wmma_f32_16x16x32_bf16 with
// pre-packed bf16 weights in WMMA B-register layout; edges: VALU + f32 atomics.
// ---------------------------------------------------------------------------

#define PI_F 3.14159265358979323846f
#define R_CUT_F 5.0f

typedef __attribute__((ext_vector_type(16))) __bf16 v16bf;
typedef __attribute__((ext_vector_type(8)))  float  v8f;

__device__ __forceinline__ float silu_f(float x) {
  return x * (1.0f / (1.0f + __expf(-x)));
}

// ---------------------------------------------------------------------------
// Pack f32 weight W (K x Nc, row-major) into bf16 WMMA B-fragment layout:
// slab = (nt * K/32 + ks), 512 bf16 per slab; within a slab lane L holds
// 16 bf16 at offset L*16: column n = nt*16 + (L&15), K = ks*32+(L>>4)*16+i.
// ---------------------------------------------------------------------------
__global__ void pack_w_kernel(const float* __restrict__ W,
                              __bf16* __restrict__ Wp, int K, int Nc) {
  const int tid   = blockIdx.x * blockDim.x + threadIdx.x;
  const int KS    = K >> 5;
  const int total = (Nc >> 4) * KS * 32;
  if (tid >= total) return;
  const int lane = tid & 31;
  const int slab = tid >> 5;
  const int nt = slab / KS, ks = slab % KS;
  const int n     = (nt << 4) + (lane & 15);
  const int kbase = (ks << 5) + ((lane >> 4) << 4);
  v16bf tmp;
#pragma unroll
  for (int i = 0; i < 16; ++i)
    tmp[i] = (__bf16)W[(size_t)(kbase + i) * Nc + n];
  *(v16bf*)(Wp + (size_t)slab * 512 + lane * 16) = tmp;
}

// ---------------------------------------------------------------------------
// C = act(A @ W + bias).  A: MxK f32 row-major; Wp: packed bf16 B-fragments.
// One wave32 per 16x64 output strip (4 WMMA tiles); K fully unrolled.
// ---------------------------------------------------------------------------
template <int K>
__global__ void wmma_gemm_bf16(const float* __restrict__ A,
                               const __bf16* __restrict__ Wp,
                               const float* __restrict__ bias,
                               float* __restrict__ C,
                               int M, int Nc, int act) {
  constexpr int KS = K / 32;
  const int lane = threadIdx.x & 31;
  const int wave = threadIdx.x >> 5;
  const int ntn4 = Nc >> 6;                       // # of 64-col strips
  const int ntm  = (M + 15) >> 4;
  const int wg   = blockIdx.x * (blockDim.x >> 5) + wave;
  if (wg >= ntm * ntn4) return;                   // wave-uniform guard

  const int mt = wg / ntn4;
  const int ng = wg % ntn4;
  const int row0 = mt << 4;
  const int lm   = lane & 15;
  const int half = lane >> 4;

  // Clamp instead of predicate: out-of-range A rows only feed output rows
  // that are masked at store time.
  int arow = row0 + lm;
  if (arow >= M) arow = M - 1;
  const float* __restrict__ Arow = A + (size_t)arow * K;
  const __bf16* __restrict__ WpL = Wp + lane * 16;

  v8f acc[4] = {};
#pragma unroll
  for (int ks = 0; ks < KS; ++ks) {
    // ---- A fragment: dwords 0-3 <- K = ks*32+half*16+[0..7],
    //                  dwords 4-7 <- K = ks*32+half*16+8+[0..7]  (2x32B) ----
    const float4* pa0 = (const float4*)(Arow + ks * 32 + half * 16);
    const float4* pa1 = (const float4*)(Arow + ks * 32 + half * 16 + 8);
    const float4 a0 = pa0[0], a1 = pa0[1];
    const float4 a2 = pa1[0], a3 = pa1[1];
    v16bf af;
    af[0]  = (__bf16)a0.x; af[1]  = (__bf16)a0.y;
    af[2]  = (__bf16)a0.z; af[3]  = (__bf16)a0.w;
    af[4]  = (__bf16)a1.x; af[5]  = (__bf16)a1.y;
    af[6]  = (__bf16)a1.z; af[7]  = (__bf16)a1.w;
    af[8]  = (__bf16)a2.x; af[9]  = (__bf16)a2.y;
    af[10] = (__bf16)a2.z; af[11] = (__bf16)a2.w;
    af[12] = (__bf16)a3.x; af[13] = (__bf16)a3.y;
    af[14] = (__bf16)a3.z; af[15] = (__bf16)a3.w;

#pragma unroll
    for (int t = 0; t < 4; ++t) {
      const int slab = (ng * 4 + t) * KS + ks;
      const v16bf bfr = *(const v16bf*)(WpL + (size_t)slab * 512);
      acc[t] = __builtin_amdgcn_wmma_f32_16x16x32_bf16(
          false, af, false, bfr, (short)0, acc[t], false, false);
    }
  }

  // ---- epilogue: VGPR r -> M = row0 + r + 8*half, N = col0 + lm ----------
  const bool full = (row0 + 16 <= M);
#pragma unroll
  for (int t = 0; t < 4; ++t) {
    const int n = (ng * 4 + t) * 16 + lm;
    const float bv = bias ? bias[n] : 0.0f;
    if (full) {
#pragma unroll
      for (int r = 0; r < 8; ++r) {
        const int m = row0 + r + (half << 3);
        float x = acc[t][r] + bv;
        if (act) x = silu_f(x);
        C[(size_t)m * Nc + n] = x;
      }
    } else {
#pragma unroll
      for (int r = 0; r < 8; ++r) {
        const int m = row0 + r + (half << 3);
        if (m < M) {
          float x = acc[t][r] + bv;
          if (act) x = silu_f(x);
          C[(size_t)m * Nc + n] = x;
        }
      }
    }
  }
}

// ---------------------------------------------------------------------------
// s[i] = emb[z[i]], v[i] = 0
// ---------------------------------------------------------------------------
__global__ void embed_init_kernel(const int* __restrict__ z,
                                  const float* __restrict__ emb,
                                  float* __restrict__ s,
                                  float* __restrict__ v, int N) {
  const int i = blockIdx.x;
  const int c = threadIdx.x;      // 0..127
  if (i >= N) return;
  s[(size_t)i * 128 + c] = emb[(size_t)z[i] * 128 + c];
#pragma unroll
  for (int k = 0; k < 3; ++k)
    v[((size_t)i * 3 + k) * 128 + c] = 0.0f;
}

__global__ void zero_f32_kernel(float* __restrict__ p, long n) {
  long i = (long)blockIdx.x * blockDim.x + threadIdx.x;
  if (i < n) p[i] = 0.0f;
}

// ---------------------------------------------------------------------------
// Edge message pass: w = (rbf@rbf_W + rbf_b)*fc, pass = w * phi[src];
// scatter d_s -> s[dst], delta_v -> v[dst].  128 threads per edge.
// ---------------------------------------------------------------------------
__global__ void edge_kernel(const int* __restrict__ edges,
                            const float* __restrict__ r_ij,
                            const float* __restrict__ rnorm,
                            const float* __restrict__ rbfW,   // 20 x 384
                            const float* __restrict__ rbfB,   // 384
                            const float* __restrict__ phi,    // N x 384
                            const float* __restrict__ v_old,  // N x 3 x 128
                            float* __restrict__ s,            // N x 128
                            float* __restrict__ v,            // N x 3 x 128
                            int E) {
  const int e = blockIdx.x;
  if (e >= E) return;
  const int c = threadIdx.x;      // 0..127

  __shared__ float sh_rbf[20];
  const int dst = edges[2 * e];
  const int src = edges[2 * e + 1];
  const float r = r_ij[e];

  if (c < 20) {
    const float n = (float)(c + 1);
    sh_rbf[c] = __sinf(n * (PI_F / R_CUT_F) * r) / r;
  }
  __syncthreads();

  const float fc = 0.5f * (__cosf(PI_F * r / R_CUT_F) + 1.0f);

  float w0 = rbfB[c], w1 = rbfB[128 + c], w2 = rbfB[256 + c];
#pragma unroll
  for (int n = 0; n < 20; ++n) {
    const float f = sh_rbf[n];
    const float* Wn = rbfW + (size_t)n * 384;
    w0 += f * Wn[c];
    w1 += f * Wn[128 + c];
    w2 += f * Wn[256 + c];
  }
  w0 *= fc; w1 *= fc; w2 *= fc;

  const float* ph = phi + (size_t)src * 384;
  const float gate = w0 * ph[c];          // dv_gate
  const float ds   = w1 * ph[128 + c];    // d_s
  const float drep = w2 * ph[256 + c];    // d_rep

  atomicAdd(&s[(size_t)dst * 128 + c], ds);
#pragma unroll
  for (int k = 0; k < 3; ++k) {
    const float dv = v_old[((size_t)src * 3 + k) * 128 + c] * gate +
                     rnorm[3 * e + k] * drep;
    atomicAdd(&v[((size_t)dst * 3 + k) * 128 + c], dv);
  }
}

// ---------------------------------------------------------------------------
// cat[i] = [ ||V[i,:,c]||_2 , s[i,c] ]
// ---------------------------------------------------------------------------
__global__ void vnorm_concat_kernel(const float* __restrict__ V,
                                    const float* __restrict__ s,
                                    float* __restrict__ cat, int N) {
  const int i = blockIdx.x;
  const int c = threadIdx.x;      // 0..127
  if (i >= N) return;
  float sum = 0.0f;
#pragma unroll
  for (int k = 0; k < 3; ++k) {
    const float x = V[((size_t)i * 3 + k) * 128 + c];
    sum += x * x;
  }
  cat[(size_t)i * 256 + c]       = sqrtf(sum);
  cat[(size_t)i * 256 + 128 + c] = s[(size_t)i * 128 + c];
}

// ---------------------------------------------------------------------------
// v += avv*U ; s += sum_k(U*V)*asv + ass
// ---------------------------------------------------------------------------
__global__ void update_final_kernel(const float* __restrict__ a,   // N x 384
                                    const float* __restrict__ U,   // N x 3 x 128
                                    const float* __restrict__ V,   // N x 3 x 128
                                    float* __restrict__ s,
                                    float* __restrict__ v, int N) {
  const int i = blockIdx.x;
  const int c = threadIdx.x;      // 0..127
  if (i >= N) return;
  const float avv = a[(size_t)i * 384 + c];
  const float asv = a[(size_t)i * 384 + 128 + c];
  const float ass = a[(size_t)i * 384 + 256 + c];
  float dot = 0.0f;
#pragma unroll
  for (int k = 0; k < 3; ++k) {
    const size_t idx = ((size_t)i * 3 + k) * 128 + c;
    const float u = U[idx];
    dot += u * V[idx];
    v[idx] += avv * u;
  }
  s[(size_t)i * 128 + c] += dot * asv + ass;
}

// ---------------------------------------------------------------------------
// out[graph_idx[i]] += blue1[i] . W2 + b2    (one wave32 per node)
// ---------------------------------------------------------------------------
__global__ void blue2_reduce_kernel(const float* __restrict__ blue1,
                                    const float* __restrict__ W2,   // 128
                                    const float* __restrict__ b2,   // 1
                                    const int* __restrict__ gidx,
                                    float* __restrict__ out, int N) {
  const int lane = threadIdx.x & 31;
  const int wv   = threadIdx.x >> 5;
  const int i    = blockIdx.x * (blockDim.x >> 5) + wv;
  if (i >= N) return;
  float sum = 0.0f;
#pragma unroll
  for (int j = 0; j < 4; ++j) {
    const int c = lane + 32 * j;
    sum += blue1[(size_t)i * 128 + c] * W2[c];
  }
#pragma unroll
  for (int off = 16; off > 0; off >>= 1)
    sum += __shfl_xor(sum, off, 32);
  if (lane == 0)
    atomicAdd(&out[gidx[i]], sum + b2[0]);
}

// ---------------------------------------------------------------------------
// Host side
// ---------------------------------------------------------------------------
static void launch_pack(const float* W, __bf16* Wp, int K, int Nc,
                        hipStream_t stream) {
  const int total = (Nc / 16) * (K / 32) * 32;
  pack_w_kernel<<<(total + 255) / 256, 256, 0, stream>>>(W, Wp, K, Nc);
}

static void launch_gemm(const float* A, const __bf16* Wp, const float* bias,
                        float* C, int M, int K, int Nc, int act,
                        hipStream_t stream) {
  const int strips = ((M + 15) / 16) * (Nc / 64);
  const int blocks = (strips + 7) / 8;         // 8 waves / 256-thread block
  if (K == 128)
    wmma_gemm_bf16<128><<<blocks, 256, 0, stream>>>(A, Wp, bias, C, M, Nc, act);
  else
    wmma_gemm_bf16<256><<<blocks, 256, 0, stream>>>(A, Wp, bias, C, M, Nc, act);
}

extern "C" void kernel_launch(void* const* d_in, const int* in_sizes, int n_in,
                              void* d_out, int out_size, void* d_ws, size_t ws_size,
                              hipStream_t stream) {
  (void)n_in; (void)ws_size;

  const int*   z        = (const int*)  d_in[0];
  const int*   edges    = (const int*)  d_in[1];
  const float* r_ij     = (const float*)d_in[2];
  const float* rnorm    = (const float*)d_in[3];
  const int*   gidx     = (const int*)  d_in[4];
  /* d_in[5] = n_graphs scalar (G == out_size) */
  const float* emb      = (const float*)d_in[6];
  const float* msg_W1   = (const float*)d_in[7];
  const float* msg_b1   = (const float*)d_in[8];
  const float* msg_W2   = (const float*)d_in[9];
  const float* msg_b2   = (const float*)d_in[10];
  const float* rbf_W    = (const float*)d_in[11];
  const float* rbf_b    = (const float*)d_in[12];
  const float* upd_UW   = (const float*)d_in[13];
  const float* upd_Ub   = (const float*)d_in[14];
  const float* upd_VW   = (const float*)d_in[15];
  const float* upd_Vb   = (const float*)d_in[16];
  const float* upd_W1   = (const float*)d_in[17];
  const float* upd_b1   = (const float*)d_in[18];
  const float* upd_W2   = (const float*)d_in[19];
  const float* upd_b2   = (const float*)d_in[20];
  const float* blue_W1  = (const float*)d_in[21];
  const float* blue_b1  = (const float*)d_in[22];
  const float* blue_W2  = (const float*)d_in[23];
  const float* blue_b2  = (const float*)d_in[24];

  const int N = in_sizes[0];          // 25000 nodes
  const int E = in_sizes[2];          // 400000 edges
  const int G = out_size;             // 1250 graphs
  float* out  = (float*)d_out;

  // ---- f32 workspace ----
  float* ws = (float*)d_ws;
  size_t o = 0;
  float* s    = ws + o; o += (size_t)N * 128;
  float* v    = ws + o; o += (size_t)N * 384;
  float* vold = ws + o; o += (size_t)N * 384;
  float* U    = ws + o; o += (size_t)N * 384;
  float* Vm   = ws + o; o += (size_t)N * 384;
  float* cat  = ws + o; o += (size_t)N * 256;
  float* t128 = ws + o; o += (size_t)N * 128;   // h1 / a1 / blue1
  float* t384 = ws + o; o += (size_t)N * 384;   // phi / a

  // ---- packed bf16 weights (B-fragment layout), after the f32 region ----
  __bf16* wp = (__bf16*)(ws + o);
  size_t p = 0;
  auto take = [&](size_t nelem) {
    __bf16* r = wp + p;
    p += (nelem + 255) & ~(size_t)255;          // keep 512B alignment
    return r;
  };
  __bf16 *p_mW1[3], *p_mW2[3], *p_uUW[3], *p_uVW[3], *p_uW1[3], *p_uW2[3];
  for (int b = 0; b < 3; ++b) {
    p_mW1[b] = take((size_t)128 * 128);
    p_mW2[b] = take((size_t)128 * 384);
    p_uUW[b] = take((size_t)128 * 128);
    p_uVW[b] = take((size_t)128 * 128);
    p_uW1[b] = take((size_t)256 * 128);
    p_uW2[b] = take((size_t)128 * 384);
  }
  __bf16* p_blW1 = take((size_t)128 * 128);

  // ---- pack all weights (per launch; deterministic) ----
  for (int b = 0; b < 3; ++b) {
    launch_pack(msg_W1 + (size_t)b * 128 * 128, p_mW1[b], 128, 128, stream);
    launch_pack(msg_W2 + (size_t)b * 128 * 384, p_mW2[b], 128, 384, stream);
    launch_pack(upd_UW + (size_t)b * 128 * 128, p_uUW[b], 128, 128, stream);
    launch_pack(upd_VW + (size_t)b * 128 * 128, p_uVW[b], 128, 128, stream);
    launch_pack(upd_W1 + (size_t)b * 256 * 128, p_uW1[b], 256, 128, stream);
    launch_pack(upd_W2 + (size_t)b * 128 * 384, p_uW2[b], 128, 384, stream);
  }
  launch_pack(blue_W1, p_blW1, 128, 128, stream);

  // ---- init: s = emb[z], v = 0 ----
  embed_init_kernel<<<N, 128, 0, stream>>>(z, emb, s, v, N);

  // ---- 3 interaction blocks ----
  for (int b = 0; b < 3; ++b) {
    const float* mb1 = msg_b1 + (size_t)b * 128;
    const float* mb2 = msg_b2 + (size_t)b * 384;
    const float* rW  = rbf_W  + (size_t)b * 20 * 384;
    const float* rb  = rbf_b  + (size_t)b * 384;
    const float* uUb = upd_Ub + (size_t)b * 128;
    const float* uVb = upd_Vb + (size_t)b * 128;
    const float* ub1 = upd_b1 + (size_t)b * 128;
    const float* ub2 = upd_b2 + (size_t)b * 384;

    // phi = silu(s @ mW1 + mb1) @ mW2 + mb2
    launch_gemm(s, p_mW1[b], mb1, t128, N, 128, 128, /*silu*/1, stream);
    launch_gemm(t128, p_mW2[b], mb2, t384, N, 128, 384, /*lin*/0, stream);

    // snapshot v (reference reads pre-update v[src])
    hipMemcpyAsync(vold, v, (size_t)N * 384 * sizeof(float),
                   hipMemcpyDeviceToDevice, stream);

    // edge pass: scatter d_s into s, delta_v into v
    edge_kernel<<<E, 128, 0, stream>>>(edges, r_ij, rnorm, rW, rb,
                                       t384, vold, s, v, E);

    // U = v @ uUW + uUb ; V = v @ uVW + uVb   (M = 3N rows, D = 128)
    launch_gemm(v, p_uUW[b], uUb, U,  3 * N, 128, 128, 0, stream);
    launch_gemm(v, p_uVW[b], uVb, Vm, 3 * N, 128, 128, 0, stream);

    // cat = [||V||, s]
    vnorm_concat_kernel<<<N, 128, 0, stream>>>(Vm, s, cat, N);

    // a = silu(cat @ uW1 + ub1) @ uW2 + ub2
    launch_gemm(cat, p_uW1[b], ub1, t128, N, 256, 128, 1, stream);
    launch_gemm(t128, p_uW2[b], ub2, t384, N, 128, 384, 0, stream);

    // gated update of s, v
    update_final_kernel<<<N, 128, 0, stream>>>(t384, U, Vm, s, v, N);
  }

  // ---- readout head ----
  launch_gemm(s, p_blW1, blue_b1, t128, N, 128, 128, 1, stream);
  zero_f32_kernel<<<(G + 255) / 256, 256, 0, stream>>>(out, (long)G);
  blue2_reduce_kernel<<<(N + 7) / 8, 256, 0, stream>>>(t128, blue_W2, blue_b2,
                                                       gidx, out, N);
}
